// VLM_68882685493677
// MI455X (gfx1250) — compile-verified, matrix operand order checked by reading, not run on hardware
//
#include <hip/hip_runtime.h>
#include <hip/hip_bf16.h>
#include <math.h>

// ---------------- model constants ----------------
#define BATCH   32
#define TTXT    64
#define EMB     768
#define NHEAD   12
#define HD      64
#define NLAYER  12
#define NCLS    256
#define NPATCH  196
#define SEQ     261          // real tokens
#define SP      272          // padded to 17*16 for WMMA tiling
#define PREFIX  64
#define FF      3072         // 4*EMB
#define QKVN    2304         // 3*EMB
#define QKVROWS 288          // qkv rows per batch, padded to K-step multiple (rows 272..287 zero)
#define ATTK    288          // att row length, padded to K-step multiple (cols >= lim zero)

typedef _Float16 half_t;
typedef __attribute__((ext_vector_type(16))) _Float16 v16h;
typedef __attribute__((ext_vector_type(8)))  _Float16 v8h;
typedef __attribute__((ext_vector_type(8)))  float    v8f;

// LDS tiles: A rows and B columns both stored k-contiguous with the same stride,
// so fragment loads are two aligned 16B LDS reads (ds_load_b128) for A and B alike.
#define LDS_STRIDE 40        // 32 K halves + 8 pad (80B rows, keeps 16B alignment)

// WMMA 16x16x32 f16 fragment: per-lane elements are k = [hi*8, hi*8+8) and
// [16+hi*8, 16+hi*8+8) of the row/column -> two contiguous 16B chunks.
__device__ __forceinline__ v16h load_frag(const half_t* tile, int idx0, int lane) {
  int l = lane & 15, hi = lane >> 4;
  const half_t* rp = tile + (idx0 + l) * LDS_STRIDE + hi * 8;
  v8h lo = *(const v8h*)(rp);
  v8h hh = *(const v8h*)(rp + 16);
  return __builtin_shufflevector(lo, hh, 0, 1, 2, 3, 4, 5, 6, 7,
                                         8, 9, 10, 11, 12, 13, 14, 15);
}

// Branch-free tile stager. Preconditions guaranteed by the launcher:
//  - K % 32 == 0 and every A row / B row has >= K valid (or zero-padded) halves
//  - M/N edges handled by index clamping; clamped lanes stage real finite data
//    whose products land only in outputs the epilogue masks.
__device__ __forceinline__ void stage_tile(half_t* As, half_t* Bt,
                                           const half_t* __restrict__ A, long long aOff, int lda, int M,
                                           const half_t* __restrict__ Bm, long long bOff, int ldb, int N,
                                           int transB, int mBase, int nBase, int kb, int tid, bool pf) {
  // A: 64 rows x 32 K, one 8-half (16B) chunk per thread
  int arow = tid >> 2, akc0 = (tid & 3) * 8;
  int agr = mBase + arow; if (agr >= M) agr = M - 1;
  const half_t* asrc = A + aOff + (long long)agr * lda + (kb + akc0);
  *(v8h*)(&As[arow * LDS_STRIDE + akc0]) = *(const v8h*)asrc;
  if (pf) __builtin_prefetch(asrc + 64, 0, 1);   // global_prefetch_b8, K-slab after next
  // B: 128 cols x 32 K, one 16-half chunk per thread (two 16B pieces)
  int bn = tid >> 1, bk0 = (tid & 1) * 16;
  int bgn = nBase + bn; if (bgn >= N) bgn = N - 1;
  if (transB) {
    const half_t* bsrc = Bm + bOff + (long long)bgn * ldb + (kb + bk0);
    *(v8h*)(&Bt[bn * LDS_STRIDE + bk0])     = *(const v8h*)(bsrc);
    *(v8h*)(&Bt[bn * LDS_STRIDE + bk0 + 8]) = *(const v8h*)(bsrc + 8);
    if (pf) __builtin_prefetch(bsrc + 64, 0, 1);
  } else {
    const half_t* bsrc = Bm + bOff + bgn;
    __align__(16) half_t tb[16];
#pragma unroll
    for (int j = 0; j < 16; ++j) tb[j] = bsrc[(long long)(kb + bk0 + j) * ldb];
    *(v8h*)(&Bt[bn * LDS_STRIDE + bk0])     = *(const v8h*)(tb);
    *(v8h*)(&Bt[bn * LDS_STRIDE + bk0 + 8]) = *(const v8h*)(tb + 8);
  }
}

// ---------------- generic batched/strided WMMA GEMM ----------------
// C[M,N] = alpha * A[M,K] @ B (+bias)(+resid)(relu?)  -> f32 and/or f16 out
// Block: 256 threads = 8 wave32 waves, 2x4 wave grid, wave tile 32x32.
// Block tile: BM=64, BN=128, BK=32, double-buffered LDS (one barrier / K-step).
// transB=1: B stored [N][K] (pre-transposed weights + K-matrix in QK^T)
// transB=0: B stored [K][N] (only att@V uses this; N=64)
__global__ __launch_bounds__(256)
void gemm_wmma_f16(const half_t* __restrict__ A, long long sAb, long long sAh, int lda,
                   const half_t* __restrict__ Bm, long long sBb, long long sBh, int ldb, int transB,
                   const float* __restrict__ bias,
                   const float* __restrict__ resid, long long sRb, long long sRh, int ldr,
                   float* __restrict__ Cf, half_t* __restrict__ Ch,
                   long long sCb, long long sChh, int ldc,
                   int M, int N, int K, int nH, float alpha, int relu) {
  __shared__ __align__(16) half_t As[2][64 * LDS_STRIDE];
  __shared__ __align__(16) half_t Bt[2][128 * LDS_STRIDE];

  int tid  = threadIdx.x;
  int lane = tid & 31;
  int wave = tid >> 5;
  int wm = wave >> 2, wn = wave & 3;       // 2 x 4 wave grid
  int rowBase = wm * 32, colBase = wn * 32;

  int z  = blockIdx.z;
  int bb = z / nH, hh = z % nH;
  long long aOff = (long long)bb * sAb + (long long)hh * sAh;
  long long bOff = (long long)bb * sBb + (long long)hh * sBh;
  long long cOff = (long long)bb * sCb + (long long)hh * sChh;
  long long rOff = (long long)bb * sRb + (long long)hh * sRh;

  int mBase = blockIdx.y * 64;
  int nBase = blockIdx.x * 128;

  v8f acc[2][2];
  acc[0][0] = {}; acc[0][1] = {}; acc[1][0] = {}; acc[1][1] = {};

  stage_tile(As[0], Bt[0], A, aOff, lda, M, Bm, bOff, ldb, N, transB,
             mBase, nBase, 0, tid, (32 < K));
  __syncthreads();

  int buf = 0;
  for (int kb = 0; kb < K; kb += 32) {
    int nkb = kb + 32;
    if (nkb < K)
      stage_tile(As[buf ^ 1], Bt[buf ^ 1], A, aOff, lda, M, Bm, bOff, ldb, N, transB,
                 mBase, nBase, nkb, tid, (nkb + 32 < K));

    v16h a0 = load_frag(As[buf], rowBase + 0,  lane);
    v16h a1 = load_frag(As[buf], rowBase + 16, lane);
    v16h b0 = load_frag(Bt[buf], colBase + 0,  lane);
    v16h b1 = load_frag(Bt[buf], colBase + 16, lane);

    acc[0][0] = __builtin_amdgcn_wmma_f32_16x16x32_f16(false, a0, false, b0, (short)0, acc[0][0], false, false);
    acc[0][1] = __builtin_amdgcn_wmma_f32_16x16x32_f16(false, a0, false, b1, (short)0, acc[0][1], false, false);
    acc[1][0] = __builtin_amdgcn_wmma_f32_16x16x32_f16(false, a1, false, b0, (short)0, acc[1][0], false, false);
    acc[1][1] = __builtin_amdgcn_wmma_f32_16x16x32_f16(false, a1, false, b1, (short)0, acc[1][1], false, false);
    __syncthreads();
    buf ^= 1;
  }

  // ---- epilogue: C/D layout -> VGPR r holds M = r + hi*8, N = lane&15 ----
  int l = lane & 15, hi = lane >> 4;
  bool full = (mBase + 64 <= M) && (nBase + 128 <= N);
#pragma unroll
  for (int ti = 0; ti < 2; ++ti) {
#pragma unroll
    for (int tj = 0; tj < 2; ++tj) {
      v8f c = acc[ti][tj];
      int gc = nBase + colBase + tj * 16 + l;
      float bval = (bias && (full || gc < N)) ? bias[gc] : 0.f;
#pragma unroll
      for (int r8 = 0; r8 < 8; ++r8) {
        int gr = mBase + rowBase + ti * 16 + hi * 8 + r8;
        if (full || (gr < M && gc < N)) {
          float v = c[r8] * alpha + bval;
          if (resid) v += resid[rOff + (long long)gr * ldr + gc];
          if (relu)  v = fmaxf(v, 0.f);
          long long ci = cOff + (long long)gr * ldc + gc;
          if (Cf) Cf[ci] = v;
          if (Ch) Ch[ci] = (half_t)v;
        }
      }
    }
  }
}

// ---------------- LayerNorm: f32 in -> f16 out, one row per block ----------------
__global__ __launch_bounds__(256)
void ln_kernel(const float* __restrict__ x, const float* __restrict__ g,
               const float* __restrict__ b, half_t* __restrict__ out, int E) {
  __shared__ float red[256];
  long long row = blockIdx.x;
  const float* xr = x + row * E;
  int tid = threadIdx.x;

  float s = 0.f;
  for (int i = tid; i < E; i += 256) s += xr[i];
  red[tid] = s; __syncthreads();
  for (int st = 128; st > 0; st >>= 1) { if (tid < st) red[tid] += red[tid + st]; __syncthreads(); }
  float mean = red[0] / E; __syncthreads();

  float v = 0.f;
  for (int i = tid; i < E; i += 256) { float d = xr[i] - mean; v += d * d; }
  red[tid] = v; __syncthreads();
  for (int st = 128; st > 0; st >>= 1) { if (tid < st) red[tid] += red[tid + st]; __syncthreads(); }
  float rstd = rsqrtf(red[0] / E + 1e-5f);

  for (int i = tid; i < E; i += 256)
    out[row * E + i] = (half_t)((xr[i] - mean) * rstd * g[i] + b[i]);
}

// ---------------- masked softmax: wei f32 (ld SP) -> att f16 (ld ATTK, zero padded) ----------------
__global__ __launch_bounds__(256)
void softmax_kernel(const float* __restrict__ wei, half_t* __restrict__ att) {
  __shared__ float red[256];
  int r = blockIdx.x;                 // query row (padded index)
  long long bh = blockIdx.y;
  const float* wr = wei + (bh * SP + r) * (long long)SP;
  half_t* ar = att + (bh * SP + r) * (long long)ATTK;
  int tid = threadIdx.x;

  if (r >= SEQ) {                     // padded query row: clean zeros
    for (int c = tid; c < ATTK; c += 256) ar[c] = (half_t)0.f;
    return;
  }
  int lim = (r >= PREFIX) ? (r + 1) : SEQ;   // valid key columns [0, lim)

  float m = -1e30f;
  for (int c = tid; c < lim; c += 256) m = fmaxf(m, wr[c]);
  red[tid] = m; __syncthreads();
  for (int st = 128; st > 0; st >>= 1) { if (tid < st) red[tid] = fmaxf(red[tid], red[tid + st]); __syncthreads(); }
  m = red[0]; __syncthreads();

  float s = 0.f;
  for (int c = tid; c < lim; c += 256) s += __expf(wr[c] - m);
  red[tid] = s; __syncthreads();
  for (int st = 128; st > 0; st >>= 1) { if (tid < st) red[tid] += red[tid + st]; __syncthreads(); }
  float inv = 1.f / red[0];

  for (int c = tid; c < ATTK; c += 256)
    ar[c] = (half_t)((c < lim) ? __expf(wr[c] - m) * inv : 0.f);
}

// ---------------- weight cast + transpose: src [L][K][N] f32 -> dst [L][N][K] f16 ----------------
__global__ void transpose_cast(const float* __restrict__ src, half_t* __restrict__ dst,
                               int L, int K, int N) {
  long long n = (long long)L * K * N;
  long long i = (long long)blockIdx.x * 256 + threadIdx.x;
  if (i >= n) return;
  long long perL = (long long)K * N;
  int l = (int)(i / perL);
  long long r = i % perL;
  int nn = (int)(r / K), kk = (int)(r % K);
  dst[i] = (half_t)src[(long long)l * perL + (long long)kk * N + nn];
}

// Wq/Wk/Wv [L,H,E,HD] -> fused transposed [L][ q|k|v x head-major n (2304) ][ E ]
__global__ void pack_qkv(const float* __restrict__ Wq, const float* __restrict__ Wk,
                         const float* __restrict__ Wv, half_t* __restrict__ dst, long long n) {
  long long i = (long long)blockIdx.x * 256 + threadIdx.x;
  if (i >= n) return;
  const long long perL = (long long)QKVN * EMB;
  int l = (int)(i / perL);
  long long r = i % perL;
  int c = (int)(r / EMB), e = (int)(r % EMB);
  int g = c / EMB, hc = c % EMB, h = hc / HD, d = hc % HD;
  const float* src = (g == 0) ? Wq : (g == 1) ? Wk : Wv;
  dst[i] = (half_t)src[(((long long)l * NHEAD + h) * EMB + e) * HD + d];
}

// ---------------- zero qkv pad rows [SP, QKVROWS) once; never rewritten ----------------
__global__ void clear_qkv_pad(half_t* __restrict__ qkv, long long n) {
  long long i = (long long)blockIdx.x * 256 + threadIdx.x;
  if (i >= n) return;
  const long long perB = (long long)(QKVROWS - SP) * QKVN;
  int b = (int)(i / perB);
  long long r = i % perB;
  long long row = SP + r / QKVN, col = r % QKVN;
  qkv[((long long)b * QKVROWS + row) * QKVN + col] = (half_t)0.f;
}

// ---------------- im2col of 16x16x3 patches -> f16 [B,196,768] ----------------
__global__ void im2col_kernel(const float* __restrict__ img, half_t* __restrict__ out, long long n) {
  long long i = (long long)blockIdx.x * 256 + threadIdx.x;
  if (i >= n) return;
  int b = (int)(i / (NPATCH * EMB));
  long long r = i % (NPATCH * EMB);
  int p = (int)(r / EMB), q = (int)(r % EMB);
  int py = p / 14, px = p % 14;
  int p1 = q / 48, rr = q % 48, p2 = rr / 3, c = rr % 3;
  out[i] = (half_t)img[(((long long)b * 224 + py * 16 + p1) * 224 + (px * 16 + p2)) * 3 + c];
}

// ---------------- assemble x: +pos, splice text & class token, zero padding ----------------
__global__ void assemble_kernel(float* __restrict__ x, const float* __restrict__ xt,
                                const float* __restrict__ ct, long long n) {
  long long i = (long long)blockIdx.x * 256 + threadIdx.x;
  if (i >= n) return;
  int b = (int)(i / ((long long)SP * EMB));
  long long r = i % ((long long)SP * EMB);
  int s = (int)(r / EMB), e = (int)(r % EMB);
  if (s >= SEQ) { x[i] = 0.f; return; }
  float ex = (float)(e & ~1) / (float)EMB;
  float freq = powf(10000.f, ex);
  float pos = (e & 1) ? cosf((float)s / freq) : sinf((float)s / freq);
  if (s < NPATCH)             x[i] = x[i] + pos;                                   // patch GEMM output
  else if (s < NPATCH + TTXT) x[i] = xt[((long long)b * TTXT + (s - NPATCH)) * EMB + e] + pos;
  else                        x[i] = ct[e] + pos;                                  // class token (s==260)
}

// ---------------- classifier head + loss ----------------
__global__ __launch_bounds__(256)
void cls_kernel(const float* __restrict__ x, const float* __restrict__ Wmlp,
                const float* __restrict__ bmlp, float* __restrict__ logits) {
  int b = blockIdx.x, a = threadIdx.x;          // NCLS == 256 == blockDim
  const float* xr = x + ((long long)b * SP + (SEQ - 1)) * EMB;
  float acc = bmlp[a];
  for (int e = 0; e < EMB; ++e) acc += xr[e] * Wmlp[(long long)e * NCLS + a];
  logits[(long long)b * NCLS + a] = acc;
}

__global__ void loss_kernel(const float* __restrict__ logits, const int* __restrict__ tgt,
                            float* __restrict__ loss) {
  __shared__ float red[BATCH];
  int b = threadIdx.x;                           // 32 threads
  const float* lr = logits + (long long)b * NCLS;
  float m = -1e30f;
  for (int a = 0; a < NCLS; ++a) m = fmaxf(m, lr[a]);
  float s = 0.f;
  for (int a = 0; a < NCLS; ++a) s += __expf(lr[a] - m);
  red[b] = lr[tgt[b]] - m - __logf(s);
  __syncthreads();
  if (b == 0) {
    float t = 0.f;
    for (int i = 0; i < BATCH; ++i) t += red[i];
    *loss = -t / (float)BATCH;
  }
}

// ---------------- host orchestration ----------------
static inline void* bump(char* base, size_t& off, size_t bytes) {
  void* p = base + off;
  off += (bytes + 255) & ~(size_t)255;
  return p;
}

extern "C" void kernel_launch(void* const* d_in, const int* in_sizes, int n_in,
                              void* d_out, int out_size, void* d_ws, size_t ws_size,
                              hipStream_t stream) {
  (void)in_sizes; (void)n_in; (void)out_size; (void)ws_size;
  const float* X_text  = (const float*)d_in[0];
  const float* X_img   = (const float*)d_in[1];
  const int*   targets = (const int*)  d_in[2];
  const float* W_patch = (const float*)d_in[3];
  const float* cls_tok = (const float*)d_in[4];
  const float* ln1_g   = (const float*)d_in[5];
  const float* ln1_b   = (const float*)d_in[6];
  const float* Wq      = (const float*)d_in[7];
  const float* Wk      = (const float*)d_in[8];
  const float* Wv      = (const float*)d_in[9];
  const float* Wo      = (const float*)d_in[10];
  const float* bo      = (const float*)d_in[11];
  const float* ln2_g   = (const float*)d_in[12];
  const float* ln2_b   = (const float*)d_in[13];
  const float* W1      = (const float*)d_in[14];
  const float* b1      = (const float*)d_in[15];
  const float* W2      = (const float*)d_in[16];
  const float* b2      = (const float*)d_in[17];
  const float* Wmlp    = (const float*)d_in[18];
  const float* bmlp    = (const float*)d_in[19];
  float* out = (float*)d_out;                    // [B*NCLS logits | loss]

  char* ws = (char*)d_ws; size_t off = 0;
  float*  x       = (float*) bump(ws, off, sizeof(float)  * BATCH * SP * EMB);
  half_t* xn      = (half_t*)bump(ws, off, sizeof(half_t) * BATCH * SP * EMB);
  half_t* qkv     = (half_t*)bump(ws, off, sizeof(half_t) * BATCH * QKVROWS * QKVN);
  float*  wei     = (float*) bump(ws, off, sizeof(float)  * (size_t)BATCH * NHEAD * SP * SP);
  half_t* att     = (half_t*)bump(ws, off, sizeof(half_t) * (size_t)BATCH * NHEAD * SP * ATTK);
  half_t* amerged = (half_t*)bump(ws, off, sizeof(half_t) * BATCH * SP * EMB);
  half_t* hbuf    = (half_t*)bump(ws, off, sizeof(half_t) * BATCH * SP * FF);
  half_t* pcol    = (half_t*)bump(ws, off, sizeof(half_t) * BATCH * NPATCH * EMB);
  half_t* WpatchT = (half_t*)bump(ws, off, sizeof(half_t) * EMB * EMB);
  half_t* WqkvT   = (half_t*)bump(ws, off, sizeof(half_t) * (size_t)NLAYER * QKVN * EMB);
  half_t* WoT     = (half_t*)bump(ws, off, sizeof(half_t) * (size_t)NLAYER * EMB * EMB);
  half_t* W1T     = (half_t*)bump(ws, off, sizeof(half_t) * (size_t)NLAYER * FF * EMB);
  half_t* W2T     = (half_t*)bump(ws, off, sizeof(half_t) * (size_t)NLAYER * EMB * FF);

  auto blocks1d = [](long long n) { return dim3((unsigned)((n + 255) / 256)); };

  // ---- weight casts (all transposed to [N][K] so GEMMs stream contiguous K) ----
  transpose_cast<<<blocks1d((long long)EMB * EMB), 256, 0, stream>>>(W_patch, WpatchT, 1, EMB, EMB);
  transpose_cast<<<blocks1d((long long)NLAYER * EMB * EMB), 256, 0, stream>>>(Wo, WoT, NLAYER, EMB, EMB);
  transpose_cast<<<blocks1d((long long)NLAYER * EMB * FF), 256, 0, stream>>>(W1, W1T, NLAYER, EMB, FF);
  transpose_cast<<<blocks1d((long long)NLAYER * FF * EMB), 256, 0, stream>>>(W2, W2T, NLAYER, FF, EMB);
  { long long n = (long long)NLAYER * QKVN * EMB;
    pack_qkv<<<blocks1d(n), 256, 0, stream>>>(Wq, Wk, Wv, WqkvT, n); }
  { long long n = (long long)BATCH * (QKVROWS - SP) * QKVN;
    clear_qkv_pad<<<blocks1d(n), 256, 0, stream>>>(qkv, n); }

  // ---- patch embed ----
  { long long n = (long long)BATCH * NPATCH * EMB;
    im2col_kernel<<<blocks1d(n), 256, 0, stream>>>(X_img, pcol, n); }
  gemm_wmma_f16<<<dim3((EMB + 127) / 128, (NPATCH + 63) / 64, BATCH), 256, 0, stream>>>(
      pcol, (long long)NPATCH * EMB, 0, EMB,
      WpatchT, 0, 0, EMB, 1,
      nullptr, nullptr, 0, 0, 0,
      x, nullptr, (long long)SP * EMB, 0, EMB,
      NPATCH, EMB, EMB, 1, 1.f, 0);
  { long long n = (long long)BATCH * SP * EMB;
    assemble_kernel<<<blocks1d(n), 256, 0, stream>>>(x, X_text, cls_tok, n); }

  const float scale = rsqrtf((float)EMB);              // reference scales by E^-0.5
  for (int l = 0; l < NLAYER; ++l) {
    // LN1
    ln_kernel<<<BATCH * SP, 256, 0, stream>>>(x, ln1_g + l * EMB, ln1_b + l * EMB, xn, EMB);
    // fused QKV: [B, 272(+16 zero pad rows), 2304] f16
    gemm_wmma_f16<<<dim3(QKVN / 128, (SP + 63) / 64, BATCH), 256, 0, stream>>>(
        xn, (long long)SP * EMB, 0, EMB,
        WqkvT + (long long)l * QKVN * EMB, 0, 0, EMB, 1,
        nullptr, nullptr, 0, 0, 0,
        nullptr, qkv, (long long)QKVROWS * QKVN, 0, QKVN,
        SP, QKVN, EMB, 1, 1.f, 0);
    // scores = scale * Q @ K^T  (per b,h); K-matrix rows are n -> transB=1
    gemm_wmma_f16<<<dim3((SP + 127) / 128, (SP + 63) / 64, BATCH * NHEAD), 256, 0, stream>>>(
        qkv, (long long)QKVROWS * QKVN, HD, QKVN,
        qkv + EMB, (long long)QKVROWS * QKVN, HD, QKVN, 1,
        nullptr, nullptr, 0, 0, 0,
        wei, nullptr, (long long)NHEAD * SP * SP, (long long)SP * SP, SP,
        SP, SP, HD, NHEAD, scale, 0);
    // masked softmax -> att f16 (ld ATTK, zero-padded tail)
    softmax_kernel<<<dim3(SP, BATCH * NHEAD), 256, 0, stream>>>(wei, att);
    // att @ V -> merged heads [B, SP, 768]; V is [K][N] inside qkv -> transB=0
    // K=ATTK=288: att cols [lim,288) are zero, qkv rows [272,288) are zero
    gemm_wmma_f16<<<dim3(1, (SP + 63) / 64, BATCH * NHEAD), 256, 0, stream>>>(
        att, (long long)NHEAD * SP * ATTK, (long long)SP * ATTK, ATTK,
        qkv + 2 * EMB, (long long)QKVROWS * QKVN, HD, QKVN, 0,
        nullptr, nullptr, 0, 0, 0,
        nullptr, amerged, (long long)SP * EMB, HD, EMB,
        SP, HD, ATTK, NHEAD, 1.f, 0);
    // out projection + bias + residual (in-place on x, f32)
    gemm_wmma_f16<<<dim3(EMB / 128, (SP + 63) / 64, BATCH), 256, 0, stream>>>(
        amerged, (long long)SP * EMB, 0, EMB,
        WoT + (long long)l * EMB * EMB, 0, 0, EMB, 1,
        bo + l * EMB,
        x, (long long)SP * EMB, 0, EMB,
        x, nullptr, (long long)SP * EMB, 0, EMB,
        SP, EMB, EMB, 1, 1.f, 0);
    // LN2
    ln_kernel<<<BATCH * SP, 256, 0, stream>>>(x, ln2_g + l * EMB, ln2_b + l * EMB, xn, EMB);
    // MLP up + ReLU
    gemm_wmma_f16<<<dim3(FF / 128, (SP + 63) / 64, BATCH), 256, 0, stream>>>(
        xn, (long long)SP * EMB, 0, EMB,
        W1T + (long long)l * FF * EMB, 0, 0, EMB, 1,
        b1 + l * FF,
        nullptr, 0, 0, 0,
        nullptr, hbuf, (long long)SP * FF, 0, FF,
        SP, FF, EMB, 1, 1.f, 1);
    // MLP down + bias + residual (in-place on x)
    gemm_wmma_f16<<<dim3(EMB / 128, (SP + 63) / 64, BATCH), 256, 0, stream>>>(
        hbuf, (long long)SP * FF, 0, FF,
        W2T + (long long)l * EMB * FF, 0, 0, FF, 1,
        b2 + l * EMB,
        x, (long long)SP * EMB, 0, EMB,
        x, nullptr, (long long)SP * EMB, 0, EMB,
        SP, EMB, FF, 1, 1.f, 0);
  }

  // classifier on class token (row 260) + loss
  cls_kernel<<<BATCH, 256, 0, stream>>>(x, Wmlp, bmlp, out);
  loss_kernel<<<1, BATCH, 0, stream>>>(out, targets, out + (long long)BATCH * NCLS);
}
